// Model_32710470926479
// MI455X (gfx1250) — compile-verified, matrix operand order checked by reading, not run on hardware
//
#include <hip/hip_runtime.h>
#include <hip/hip_bf16.h>

// ---------------- model dims ----------------
#define BSZ    4
#define HM     128
#define WM     128
#define NTOK   65536          // BSZ*HM*WM
#define CDIM   256
#define HEADS  8
#define DDIM   32
#define GSL    64             // SLICES
#define LAYERS 5

typedef __attribute__((ext_vector_type(16))) __bf16 v16bf;
typedef __attribute__((ext_vector_type(8)))  float  v8f;
typedef __attribute__((__vector_size__(4 * sizeof(int)))) int v4i_t;

// ---------------- helpers ----------------
__device__ inline float gelu_exact(float x) {
    return 0.5f * x * (1.0f + erff(x * 0.70710678118654752440f));
}

__device__ inline unsigned short f2bf(float f) {
    unsigned u = __float_as_uint(f);
    u += 0x7FFFu + ((u >> 16) & 1u);   // round-to-nearest-even
    return (unsigned short)(u >> 16);
}

// -------- gfx1250 async global->LDS copy (ASYNCcnt) with sync fallback --------
#if __has_builtin(__builtin_amdgcn_global_load_async_to_lds_b128) && \
    __has_builtin(__builtin_amdgcn_s_wait_asynccnt)
#define USE_ASYNC 1
#else
#define USE_ASYNC 0
#endif

__device__ inline void copy16(const void* g, void* l) {
#if USE_ASYNC
    __builtin_amdgcn_global_load_async_to_lds_b128((v4i_t*)g, (v4i_t*)l, 0, 0);
#else
    *(uint4*)l = *(const uint4*)g;
#endif
}
__device__ inline void wait_async_prev() {   // leave the 4 just-issued in flight
#if USE_ASYNC
    __builtin_amdgcn_s_wait_asynccnt(4);
#endif
}
__device__ inline void wait_async_all() {
#if USE_ASYNC
    __builtin_amdgcn_s_wait_asynccnt(0);
#endif
}

// Fragment loader: 16x32 bf16 tile slice from LDS (row-major, stride 40 elems = 80B).
// ISA A/B layout: lanes 0-15 -> row=lane, K {0..7,16..23}; lanes 16-31 -> row=lane-16, K {8..15,24..31}.
#define LDS_STR 40
__device__ inline v16bf load_frag(const unsigned short* base, int row0) {
    union { v16bf v; unsigned u[8]; } r;
    const int lane = threadIdx.x & 31;
    const int half = lane >> 4;
    const int m    = lane & 15;
    const unsigned short* p = base + (row0 + m) * LDS_STR + half * 8;
#pragma unroll
    for (int t = 0; t < 4; ++t) {
        r.u[t]     = *(const unsigned*)(p + 2 * t);
        r.u[t + 4] = *(const unsigned*)(p + 2 * t + 16);
    }
    return r.v;
}

__device__ inline v8f wmma_bf16(v16bf a, v16bf b, v8f c) {
    return __builtin_amdgcn_wmma_f32_16x16x32_bf16(false, a, false, b, (short)0, c, false, false);
}

// ========== weight convert+transpose: WT[n*K+k] = bf16(W[k*Nc+n]) ==========
__global__ __launch_bounds__(256) void wt_kernel(
    const float* __restrict__ W, unsigned short* __restrict__ WT, int K, int Nc)
{
    const size_t idx = (size_t)blockIdx.x * 256 + threadIdx.x;
    const size_t n = idx / (size_t)K, k = idx - n * (size_t)K;
    WT[idx] = f2bf(W[k * (size_t)Nc + n]);
}

__global__ void zero_kernel(unsigned short* z) {
    if (threadIdx.x < 16) z[threadIdx.x] = 0;
}

// ================= generic GEMM: out = act(A[M,K]bf16 @ WT[N,K]bf16^T + bias) (+resid) =======
// grid.x = M/128, grid.y = Nc/128, block = 256 (8 waves, each 64x32)
template <int ACT, bool RESID, bool OUTBF>
__global__ __launch_bounds__(256) void gemm_kernel(
    const unsigned short* __restrict__ A, const unsigned short* __restrict__ Wt,
    const float* __restrict__ bias, const float* __restrict__ resid,
    void* __restrict__ outp, int K, int Nc)
{
    __shared__ unsigned short As[2][128 * LDS_STR];
    __shared__ unsigned short Bs[2][128 * LDS_STR];
    const int t    = threadIdx.x;
    const int lane = t & 31;
    const int wave = t >> 5;
    const int wm   = wave & 1;
    const int wn   = wave >> 1;
    const int m0   = blockIdx.x * 128;
    const int n0   = blockIdx.y * 128;

    const v8f z8 = {0.f,0.f,0.f,0.f,0.f,0.f,0.f,0.f};
    v8f acc[4][2];
#pragma unroll
    for (int i = 0; i < 4; ++i) { acc[i][0] = z8; acc[i][1] = z8; }

    const unsigned short* Abase = A  + (size_t)m0 * K;
    const unsigned short* Bbase = Wt + (size_t)n0 * K;

    auto stage = [&](int ks, int bid) {
        const int k0 = ks * 32;
#pragma unroll
        for (int q = 0; q < 2; ++q) {
            const int c = q * 256 + t;          // 0..511 chunk id
            const int r = c >> 2, part = c & 3; // row, 16B quarter
            copy16(Abase + (size_t)r * K + k0 + part * 8, &As[bid][r * LDS_STR + part * 8]);
            copy16(Bbase + (size_t)r * K + k0 + part * 8, &Bs[bid][r * LDS_STR + part * 8]);
        }
    };

    const int nk = K >> 5;
    stage(0, 0);
    for (int ks = 0; ks < nk; ++ks) {
        const int cur = ks & 1;
        if (ks + 1 < nk) { stage(ks + 1, cur ^ 1); wait_async_prev(); }
        else             { wait_async_all(); }
        __syncthreads();
        v16bf bf0 = load_frag(Bs[cur], wn * 32);
        v16bf bf1 = load_frag(Bs[cur], wn * 32 + 16);
#pragma unroll
        for (int i = 0; i < 4; ++i) {
            v16bf af = load_frag(As[cur], wm * 64 + i * 16);
            acc[i][0] = wmma_bf16(af, bf0, acc[i][0]);
            acc[i][1] = wmma_bf16(af, bf1, acc[i][1]);
        }
        __syncthreads();
    }

    float* outf = (float*)outp;
    unsigned short* outh = (unsigned short*)outp;
    const int half = lane >> 4, colL = lane & 15;
#pragma unroll
    for (int i = 0; i < 4; ++i) {
        const int rowb = m0 + wm * 64 + i * 16 + half * 8;
#pragma unroll
        for (int j = 0; j < 2; ++j) {
            const int col = n0 + wn * 32 + j * 16 + colL;
            const float bv = bias[col];
#pragma unroll
            for (int r = 0; r < 8; ++r) {
                float v = acc[i][j][r] + bv;
                if (ACT) v = gelu_exact(v);
                const size_t off = (size_t)(rowb + r) * Nc + col;
                if (RESID) v += resid[off];
                if (OUTBF) outh[off] = f2bf(v); else outf[off] = v;
            }
        }
    }
}

// ================= 3x3 SAME conv as im2col WMMA GEMM =================
// A = z (bf16 [B,H,W,C]), WT = weights bf16 [256n][2304k]; grid.x=B*HM, grid.y=2
__global__ __launch_bounds__(256) void conv_kernel(
    const unsigned short* __restrict__ z, const unsigned short* __restrict__ Wt,
    const float* __restrict__ bias, const unsigned short* __restrict__ zchunk,
    float* __restrict__ out)
{
    __shared__ unsigned short As[2][128 * LDS_STR];
    __shared__ unsigned short Bs[2][128 * LDS_STR];
    const int t    = threadIdx.x;
    const int lane = t & 31;
    const int wave = t >> 5;
    const int wm   = wave & 1;
    const int wn   = wave >> 1;
    const int bi   = blockIdx.x;          // b*128 + hrow
    const int b    = bi >> 7;
    const int hrow = bi & 127;
    const int n0   = blockIdx.y * 128;

    const v8f z8 = {0.f,0.f,0.f,0.f,0.f,0.f,0.f,0.f};
    v8f acc[4][2];
#pragma unroll
    for (int i = 0; i < 4; ++i) { acc[i][0] = z8; acc[i][1] = z8; }

    auto stage = [&](int ks, int bid) {
        const int idx9 = ks >> 3;           // (ky,kx), 8 k-steps each
        const int c0   = (ks & 7) * 32;
        const int ky   = idx9 / 3, kx = idx9 % 3;
        const int in_h = hrow + ky - 1;
        const bool hok = (in_h >= 0) && (in_h < HM);
#pragma unroll
        for (int q = 0; q < 2; ++q) {
            const int c = q * 256 + t;
            const int r = c >> 2, part = c & 3;
            const int in_w = r + kx - 1;
            // zero-pad via redirect to a zeroed 16B chunk (keeps ASYNCcnt uniform)
            const unsigned short* srcA =
                (hok && in_w >= 0 && in_w < WM)
                ? z + ((size_t)((b * HM + in_h) * WM + in_w) * CDIM + c0 + part * 8)
                : zchunk;
            copy16(srcA, &As[bid][r * LDS_STR + part * 8]);
            copy16(Wt + (size_t)(n0 + r) * 2304 + ks * 32 + part * 8,
                   &Bs[bid][r * LDS_STR + part * 8]);
        }
    };

    stage(0, 0);
    for (int ks = 0; ks < 72; ++ks) {
        const int cur = ks & 1;
        if (ks + 1 < 72) { stage(ks + 1, cur ^ 1); wait_async_prev(); }
        else             { wait_async_all(); }
        __syncthreads();
        v16bf bf0 = load_frag(Bs[cur], wn * 32);
        v16bf bf1 = load_frag(Bs[cur], wn * 32 + 16);
#pragma unroll
        for (int i = 0; i < 4; ++i) {
            v16bf af = load_frag(As[cur], wm * 64 + i * 16);
            acc[i][0] = wmma_bf16(af, bf0, acc[i][0]);
            acc[i][1] = wmma_bf16(af, bf1, acc[i][1]);
        }
        __syncthreads();
    }

    const int half = lane >> 4, colL = lane & 15;
#pragma unroll
    for (int i = 0; i < 4; ++i) {
        const int rowb = bi * 128 + wm * 64 + i * 16 + half * 8;
#pragma unroll
        for (int j = 0; j < 2; ++j) {
            const int col = n0 + wn * 32 + j * 16 + colL;
            const float bv = bias[col];
#pragma unroll
            for (int r = 0; r < 8; ++r)
                out[(size_t)(rowb + r) * CDIM + col] = acc[i][j][r] + bv;
        }
    }
}

// ================= LayerNorm: one wave per token, bf16 output =================
__global__ __launch_bounds__(256) void ln_kernel(
    const float* __restrict__ x, const float* __restrict__ g,
    const float* __restrict__ bb, unsigned short* __restrict__ y)
{
    const int wave = threadIdx.x >> 5, lane = threadIdx.x & 31;
    const int tok = blockIdx.x * 8 + wave;
    const float* xr = x + (size_t)tok * CDIM;
    float v[8]; float s = 0.f;
#pragma unroll
    for (int i = 0; i < 8; ++i) { v[i] = xr[lane + 32 * i]; s += v[i]; }
#pragma unroll
    for (int m = 16; m >= 1; m >>= 1) s += __shfl_xor(s, m, 32);
    const float mean = s * (1.f / 256.f);
    float vs = 0.f;
#pragma unroll
    for (int i = 0; i < 8; ++i) { float d = v[i] - mean; vs += d * d; }
#pragma unroll
    for (int m = 16; m >= 1; m >>= 1) vs += __shfl_xor(vs, m, 32);
    const float rs = rsqrtf(vs * (1.f / 256.f) + 1e-5f);
    unsigned short* yr = y + (size_t)tok * CDIM;
#pragma unroll
    for (int i = 0; i < 8; ++i) {
        const int c = lane + 32 * i;
        yr[c] = f2bf((v[i] - mean) * rs * g[c] + bb[c]);
    }
}

// ================= slice logits + softmax over G=64; one thread per (b,h,n) =================
__global__ __launch_bounds__(256) void slice_kernel(
    const float* __restrict__ xm, const float* __restrict__ sW,
    const float* __restrict__ sb, const float* __restrict__ temp,
    float* __restrict__ sw)
{
    const int idx = blockIdx.x * 256 + threadIdx.x;
    const int n  = idx & 16383;
    const int bh = idx >> 14;
    const int hh = bh & 7, b = bh >> 3;
    const float* xr = xm + ((size_t)b * 16384 + n) * CDIM + hh * DDIM;
    float xv[32];
#pragma unroll
    for (int d = 0; d < 32; ++d) xv[d] = xr[d];
    const float invt = 1.f / temp[hh];
    float mx = -1e30f;
    for (int g = 0; g < GSL; ++g) {
        float s = sb[g];
#pragma unroll
        for (int d = 0; d < 32; ++d) s += xv[d] * sW[d * GSL + g];
        mx = fmaxf(mx, s * invt);
    }
    float* dst = sw + ((size_t)bh * 16384 + n) * GSL;
    float sum = 0.f;
    for (int g = 0; g < GSL; ++g) {
        float s = sb[g];
#pragma unroll
        for (int d = 0; d < 32; ++d) s += xv[d] * sW[d * GSL + g];
        float e = __expf(s * invt - mx);
        sum += e;
        dst[g] = e;
    }
    const float inv = 1.f / sum;
    for (int g = 0; g < GSL; ++g) dst[g] *= inv;
}

// ================= tok partial: sw^T @ fx_mid over n-chunk (split-K) =================
__global__ __launch_bounds__(256) void tok_partial_kernel(
    const float* __restrict__ fxm, const float* __restrict__ sw,
    float* __restrict__ partial, float* __restrict__ pnorm)
{
    __shared__ float swt[16 * 64];
    __shared__ float fxt[16 * 32];
    const int bh = blockIdx.x, ch = blockIdx.y, t = threadIdx.x;
    const int b = bh >> 3, hh = bh & 7;
    const int n0 = ch * 512;
    const int g = t >> 2, d0 = (t & 3) * 8;
    float acc[8] = {0,0,0,0,0,0,0,0};
    float nrm = 0.f;
    for (int nb = 0; nb < 32; ++nb) {
        const int nbase = n0 + nb * 16;
#pragma unroll
        for (int q = 0; q < 4; ++q) {
            const int e = t * 4 + q, nl = e >> 6, gg = e & 63;
            swt[e] = sw[((size_t)bh * 16384 + nbase + nl) * GSL + gg];
        }
#pragma unroll
        for (int q = 0; q < 2; ++q) {
            const int e = t * 2 + q, nl = e >> 5, dd = e & 31;
            fxt[e] = fxm[((size_t)b * 16384 + nbase + nl) * CDIM + hh * DDIM + dd];
        }
        __syncthreads();
#pragma unroll
        for (int i = 0; i < 16; ++i) {
            const float a = swt[i * 64 + g];
            nrm += a;
#pragma unroll
            for (int q = 0; q < 8; ++q) acc[q] += a * fxt[i * 32 + d0 + q];
        }
        __syncthreads();
    }
    float* pp = partial + ((size_t)ch * 32 + bh) * 2048;
#pragma unroll
    for (int q = 0; q < 8; ++q) pp[g * 32 + d0 + q] = acc[q];
    if (d0 == 0) pnorm[((size_t)ch * 32 + bh) * 64 + g] = nrm;
}

__global__ __launch_bounds__(256) void tok_finalize_kernel(
    const float* __restrict__ partial, const float* __restrict__ pnorm,
    float* __restrict__ tok)
{
    const int bh = blockIdx.x, t = threadIdx.x;
    const int g = t >> 2, d0 = (t & 3) * 8;
    float acc[8] = {0,0,0,0,0,0,0,0};
    float nrm = 0.f;
    for (int ch = 0; ch < 32; ++ch) {
        const float* pp = partial + ((size_t)ch * 32 + bh) * 2048;
#pragma unroll
        for (int q = 0; q < 8; ++q) acc[q] += pp[g * 32 + d0 + q];
        nrm += pnorm[((size_t)ch * 32 + bh) * 64 + g];
    }
    const float inv = 1.f / (nrm + 1e-5f);
#pragma unroll
    for (int q = 0; q < 8; ++q) tok[(size_t)bh * 2048 + g * 32 + d0 + q] = acc[q] * inv;
}

// ================= tiny attention among 64 slice tokens, per (b,h) =================
__global__ __launch_bounds__(256) void attn_kernel(
    const float* __restrict__ tok, const float* __restrict__ qW,
    const float* __restrict__ kW, const float* __restrict__ vW,
    float* __restrict__ o)
{
    __shared__ float tk[2048], qm[2048], km[2048], vm[2048], at[64 * 64];
    const int bh = blockIdx.x, t = threadIdx.x;
    const float* tb = tok + (size_t)bh * 2048;
#pragma unroll
    for (int i = 0; i < 8; ++i) tk[t + 256 * i] = tb[t + 256 * i];
    __syncthreads();
    const int g = t >> 2, d0 = (t & 3) * 8;
#pragma unroll
    for (int q = 0; q < 8; ++q) {
        const int d = d0 + q;
        float sq = 0, sk = 0, sv = 0;
        for (int e = 0; e < 32; ++e) {
            const float tv = tk[g * 32 + e];
            sq += tv * qW[e * 32 + d];
            sk += tv * kW[e * 32 + d];
            sv += tv * vW[e * 32 + d];
        }
        qm[g * 32 + d] = sq; km[g * 32 + d] = sk; vm[g * 32 + d] = sv;
    }
    __syncthreads();
    const int c0 = (t & 3) * 16;
    for (int cc = 0; cc < 16; ++cc) {
        const int kc = c0 + cc;
        float s = 0;
        for (int e = 0; e < 32; ++e) s += qm[g * 32 + e] * km[kc * 32 + e];
        at[g * 64 + kc] = s * 0.17677669529663687f;   // 1/sqrt(32)
    }
    __syncthreads();
    if (t < 64) {
        float mx = -1e30f;
        for (int k2 = 0; k2 < 64; ++k2) mx = fmaxf(mx, at[t * 64 + k2]);
        float sum = 0;
        for (int k2 = 0; k2 < 64; ++k2) { float e = __expf(at[t * 64 + k2] - mx); at[t * 64 + k2] = e; sum += e; }
        const float inv = 1.f / sum;
        for (int k2 = 0; k2 < 64; ++k2) at[t * 64 + k2] *= inv;
    }
    __syncthreads();
    float* ob = o + (size_t)bh * 2048;
#pragma unroll
    for (int q = 0; q < 8; ++q) {
        const int d = d0 + q;
        float s = 0;
        for (int k2 = 0; k2 < 64; ++k2) s += at[g * 64 + k2] * vm[k2 * 32 + d];
        ob[g * 32 + d] = s;
    }
}

// ================= deslice: bf16 out[b,n,h*32+d] = sum_g sw*o =================
__global__ __launch_bounds__(256) void deslice_kernel(
    const float* __restrict__ sw, const float* __restrict__ o,
    unsigned short* __restrict__ outb)
{
    __shared__ float ot[2048];
    const int ch = blockIdx.x, bh = blockIdx.y, t = threadIdx.x;
    const int b = bh >> 3, hh = bh & 7;
    const float* ob = o + (size_t)bh * 2048;
#pragma unroll
    for (int q = 0; q < 8; ++q) ot[t + 256 * q] = ob[t + 256 * q];
    __syncthreads();
    const int n = ch * 256 + t;
    const float* swr = sw + ((size_t)bh * 16384 + n) * GSL;
    float acc[32];
#pragma unroll
    for (int d = 0; d < 32; ++d) acc[d] = 0.f;
    for (int g = 0; g < GSL; ++g) {
        const float a = swr[g];
#pragma unroll
        for (int d = 0; d < 32; ++d) acc[d] += a * ot[g * 32 + d];
    }
    unsigned short* dst = outb + ((size_t)b * 16384 + n) * CDIM + hh * DDIM;
#pragma unroll
    for (int d = 0; d < 32; d += 2) {
        unsigned pk = (unsigned)f2bf(acc[d]) | ((unsigned)f2bf(acc[d + 1]) << 16);
        *(unsigned*)(dst + d) = pk;
    }
}

// ================= preprocess first matmul (K=3) + gelu, bf16 out =================
__global__ __launch_bounds__(256) void pre_kernel(
    const float* __restrict__ x, const float* __restrict__ fx,
    const float* __restrict__ W1, const float* __restrict__ b1,
    unsigned short* __restrict__ h1)
{
    const size_t idx = (size_t)blockIdx.x * 256 + threadIdx.x;
    const size_t tok = idx >> 9;
    const int j = (int)(idx & 511);
    const float x0 = x[tok * 2], x1 = x[tok * 2 + 1], f = fx[tok];
    const float s = x0 * W1[j] + x1 * W1[512 + j] + f * W1[1024 + j] + b1[j];
    h1[idx] = f2bf(gelu_exact(s));
}

// ================= final head: LN3 then dot with [256,1] =================
__global__ __launch_bounds__(256) void head_kernel(
    const float* __restrict__ x, const float* __restrict__ g,
    const float* __restrict__ bb, const float* __restrict__ w,
    const float* __restrict__ wb, float* __restrict__ out)
{
    const int wave = threadIdx.x >> 5, lane = threadIdx.x & 31;
    const int tok = blockIdx.x * 8 + wave;
    const float* xr = x + (size_t)tok * CDIM;
    float v[8]; float s = 0.f;
#pragma unroll
    for (int i = 0; i < 8; ++i) { v[i] = xr[lane + 32 * i]; s += v[i]; }
#pragma unroll
    for (int m = 16; m >= 1; m >>= 1) s += __shfl_xor(s, m, 32);
    const float mean = s * (1.f / 256.f);
    float vs = 0.f;
#pragma unroll
    for (int i = 0; i < 8; ++i) { float d = v[i] - mean; vs += d * d; }
#pragma unroll
    for (int m = 16; m >= 1; m >>= 1) vs += __shfl_xor(vs, m, 32);
    const float rs = rsqrtf(vs * (1.f / 256.f) + 1e-5f);
    float acc = 0.f;
#pragma unroll
    for (int i = 0; i < 8; ++i) {
        const int c = lane + 32 * i;
        acc += ((v[i] - mean) * rs * g[c] + bb[c]) * w[c];
    }
#pragma unroll
    for (int m = 16; m >= 1; m >>= 1) acc += __shfl_xor(acc, m, 32);
    if (lane == 0) out[tok] = acc + wb[0];
}

// ============================= launcher =============================
extern "C" void kernel_launch(void* const* d_in, const int* in_sizes, int n_in,
                              void* d_out, int out_size, void* d_ws, size_t ws_size,
                              hipStream_t stream) {
    const float* x      = (const float*)d_in[0];
    const float* fx     = (const float*)d_in[1];
    const float* preW1  = (const float*)d_in[2];
    const float* preb1  = (const float*)d_in[3];
    const float* preW2  = (const float*)d_in[4];
    const float* preb2  = (const float*)d_in[5];
    const float* ln1g   = (const float*)d_in[6];
    const float* ln1b   = (const float*)d_in[7];
    const float* cxW    = (const float*)d_in[8];
    const float* cxb    = (const float*)d_in[9];
    const float* cfW    = (const float*)d_in[10];
    const float* cfb    = (const float*)d_in[11];
    const float* sliceW = (const float*)d_in[12];
    const float* sliceb = (const float*)d_in[13];
    const float* temp   = (const float*)d_in[14];
    const float* qW     = (const float*)d_in[15];
    const float* kW     = (const float*)d_in[16];
    const float* vW     = (const float*)d_in[17];
    const float* outW   = (const float*)d_in[18];
    const float* outb   = (const float*)d_in[19];
    const float* ln2g   = (const float*)d_in[20];
    const float* ln2b   = (const float*)d_in[21];
    const float* mW1    = (const float*)d_in[22];
    const float* mb1    = (const float*)d_in[23];
    const float* mW2    = (const float*)d_in[24];
    const float* mb2    = (const float*)d_in[25];
    const float* ln3g   = (const float*)d_in[26];
    const float* ln3b   = (const float*)d_in[27];
    const float* m2W    = (const float*)d_in[28];
    const float* m2b    = (const float*)d_in[29];
    float* out = (float*)d_out;
    float* ws  = (float*)d_ws;

    const size_t P = (size_t)NTOK * CDIM;   // 16,777,216 floats
    float* h       = ws;                    // residual stream f32
    float* xb      = ws + P;                // x_mid f32
    float* fb      = ws + 2 * P;            // fx_mid f32
    float* swb     = ws + 3 * P;            // sw f32 (2P)
    float* partial = ws + 5 * P;            // 32*32*2048
    float* pnorm   = partial + 32 * 32 * 2048;
    float* tokb    = pnorm + 32 * 32 * 64;
    float* ob      = tokb + 32 * 2048;

    // bf16 (ushort) region
    unsigned short* us = (unsigned short*)(ob + 32 * 2048);
    unsigned short* zchunk = us;                                   // 16 (zero pad chunk)
    unsigned short* zbbf   = zchunk + 16;                          // LN out bf16 [NTOK,256]
    unsigned short* abf    = zbbf + (size_t)NTOK * CDIM;           // [NTOK,512]: h1 / attn|mid
    unsigned short* attnbf = abf;
    unsigned short* midbf  = abf + (size_t)NTOK * CDIM;
    unsigned short* wts    = abf + (size_t)NTOK * 512;
    unsigned short* preW2T = wts;                                  // 512*256
    unsigned short* convxT = preW2T + 512 * 256;                   // 5 x 2304*256
    unsigned short* convfT = convxT + (size_t)5 * 2304 * 256;
    unsigned short* outWT  = convfT + (size_t)5 * 2304 * 256;      // 5 x 256*256
    unsigned short* mW1T   = outWT + (size_t)5 * 256 * 256;
    unsigned short* mW2T   = mW1T + (size_t)5 * 256 * 256;

    // one-time per launch: zero chunk + weight convert/transpose to bf16 [N,K]
    zero_kernel<<<1, 32, 0, stream>>>(zchunk);
    wt_kernel<<<(512 * 256) / 256, 256, 0, stream>>>(preW2, preW2T, 512, 256);
    for (int L = 0; L < LAYERS; ++L) {
        const size_t wc = (size_t)L * 2304 * 256;
        const size_t wm = (size_t)L * 256 * 256;
        wt_kernel<<<(2304 * 256) / 256, 256, 0, stream>>>(cxW + wc, convxT + wc, 2304, 256);
        wt_kernel<<<(2304 * 256) / 256, 256, 0, stream>>>(cfW + wc, convfT + wc, 2304, 256);
        wt_kernel<<<(256 * 256) / 256, 256, 0, stream>>>(outW + wm, outWT + wm, 256, 256);
        wt_kernel<<<(256 * 256) / 256, 256, 0, stream>>>(mW1 + wm, mW1T + wm, 256, 256);
        wt_kernel<<<(256 * 256) / 256, 256, 0, stream>>>(mW2 + wm, mW2T + wm, 256, 256);
    }

    // preprocess: [x|fx] -> gelu(@W1) -> @W2
    pre_kernel<<<(NTOK * 512) / 256, 256, 0, stream>>>(x, fx, preW1, preb1, abf);
    gemm_kernel<0, false, false><<<dim3(NTOK / 128, 2), 256, 0, stream>>>(
        abf, preW2T, preb2, nullptr, h, 512, 256);

    for (int L = 0; L < LAYERS; ++L) {
        const size_t wc = (size_t)L * 2304 * 256;
        const size_t wm = (size_t)L * 256 * 256;
        ln_kernel<<<NTOK / 8, 256, 0, stream>>>(h, ln1g + L * 256, ln1b + L * 256, zbbf);
        conv_kernel<<<dim3(BSZ * HM, 2), 256, 0, stream>>>(zbbf, convxT + wc, cxb + L * 256, zchunk, xb);
        conv_kernel<<<dim3(BSZ * HM, 2), 256, 0, stream>>>(zbbf, convfT + wc, cfb + L * 256, zchunk, fb);
        slice_kernel<<<(NTOK * HEADS) / 256, 256, 0, stream>>>(
            xb, sliceW + (size_t)L * DDIM * GSL, sliceb + L * GSL, temp + L * HEADS, swb);
        tok_partial_kernel<<<dim3(32, 32), 256, 0, stream>>>(fb, swb, partial, pnorm);
        tok_finalize_kernel<<<32, 256, 0, stream>>>(partial, pnorm, tokb);
        attn_kernel<<<32, 256, 0, stream>>>(tokb, qW + L * 1024, kW + L * 1024, vW + L * 1024, ob);
        deslice_kernel<<<dim3(64, 32), 256, 0, stream>>>(swb, ob, attnbf);
        gemm_kernel<0, true, false><<<dim3(NTOK / 128, 2), 256, 0, stream>>>(
            attnbf, outWT + wm, outb + L * 256, h, h, 256, 256);
        ln_kernel<<<NTOK / 8, 256, 0, stream>>>(h, ln2g + L * 256, ln2b + L * 256, zbbf);
        gemm_kernel<1, false, true><<<dim3(NTOK / 128, 2), 256, 0, stream>>>(
            zbbf, mW1T + wm, mb1 + L * 256, nullptr, midbf, 256, 256);
        gemm_kernel<0, true, false><<<dim3(NTOK / 128, 2), 256, 0, stream>>>(
            midbf, mW2T + wm, mb2 + L * 256, h, h, 256, 256);
    }

    head_kernel<<<NTOK / 8, 256, 0, stream>>>(h, ln3g, ln3b, m2W, m2b, out);

    (void)in_sizes; (void)n_in; (void)out_size; (void)ws_size;
}
